// MultiHeadAttention_54992761258674
// MI455X (gfx1250) — matrix operand order repaired and checked by reading, not compiled, not run on hardware
//
#include <hip/hip_runtime.h>

// Problem constants (match reference)
#define B_ 2
#define S_ 2048
#define E_ 1024
#define H_ 16
#define D_ 64
#define SCALE_ 0.125f  // D^-0.5

typedef _Float16 half_t;
typedef __attribute__((ext_vector_type(16))) _Float16 v16h;
typedef __attribute__((ext_vector_type(8)))  _Float16 v8h;
typedef __attribute__((ext_vector_type(8)))  float    v8f;
typedef __attribute__((ext_vector_type(4)))  unsigned int v4u;
typedef __attribute__((ext_vector_type(8)))  int      v8i;
typedef __attribute__((ext_vector_type(4)))  int      v4i;

static __device__ __forceinline__ v8f wmma_f16(v16h a, v16h b, v8f c) {
  // (neg_a, A, neg_b, B, c_mod, C, reuse_a, reuse_b)
  return __builtin_amdgcn_wmma_f32_16x16x32_f16(false, a, false, b, (short)0, c,
                                                false, false);
}

// ---- WMMA fragment loaders (wave32, 16x16x32 f16; contiguous k order: lane half h holds k = 16h..16h+15) ----

// A-fragment 16x32 (MxK): lane L holds row M=L&15; K = k0..k0+15
// with k0 = (L>>4)*16. Memory: element (m,k) at base[m*ld + k].
static __device__ __forceinline__ v16h load_frag_rowK(const half_t* base, int ld) {
  const int lane = threadIdx.x & 31;
  const int row  = lane & 15;
  const int k0   = (lane >> 4) << 4;  // 0 or 16
  const half_t* p = base + (size_t)row * ld;
  v16h f;
#pragma unroll
  for (int i = 0; i < 16; ++i) {
    f[i]     = p[k0 + i];
  }
  return f;
}

// B-fragment 32x16 (KxN) where column n is contiguous over K in memory:
// element (k,n) at base[n*ld + k]. Lane L holds column n=L&15, K = kb..kb+15.
static __device__ __forceinline__ v16h load_fragB_colK(const half_t* base, int ld) {
  const int lane = threadIdx.x & 31;
  const int col  = lane & 15;
  const int kb   = (lane >> 4) << 4;  // 0 or 16
  const half_t* p = base + (size_t)col * ld + kb;
  v16h f;
#pragma unroll
  for (int i = 0; i < 16; ++i) f[i] = p[i];
  return f;
}

// B-fragment 32x16 (KxN) where memory is row-major over K with stride ld:
// element (k,n) at base[k*ld + n].
static __device__ __forceinline__ v16h load_fragB_rowK(const half_t* base, int ld) {
  const int lane = threadIdx.x & 31;
  const int col  = lane & 15;
  const int kb   = (lane >> 4) << 4;
  v16h f;
#pragma unroll
  for (int i = 0; i < 16; ++i) f[i] = base[(size_t)(kb + i) * ld + col];
  return f;
}

// ---- Tensor Data Mover: DMA a 64-row x 32-half W tile (row stride E) to LDS.
// D# per CDNA5 ISA §8.3/8.4; 2-D tile so groups 2/3 and the extra group are zero.
// 6-arg builtin (clang-23 / therock-10.0): (v4u, v8i, v4i, v4i, v8i, i32 cpol).
static __device__ __forceinline__ void tdm_issue_w_tile(const half_t* W, int tileN,
                                                        int kk, half_t* ldsDst) {
  const unsigned lds = (unsigned)(size_t)ldsDst;   // LDS byte offset (low 32 bits)
  const unsigned long long ga =
      (unsigned long long)(size_t)(W + (size_t)tileN * E_ + kk);
  v4u g0;
  g0[0] = 1u;                         // count=1 (valid user descriptor)
  g0[1] = lds;                        // lds_addr
  g0[2] = (unsigned)ga;               // global_addr[31:0]
  g0[3] = (unsigned)((ga >> 32) & 0x01FFFFFFu) | 0x80000000u;  // [56:32] | type=2
  v8i g1;
  g1[0] = 0x00010000;                 // workgroup_mask=0, data_size=1 (2B), flags=0
  g1[1] = (int)((unsigned)E_ << 16);  // tensor_dim0[15:0]=E (atomic_bar_addr=0)
  g1[2] = (int)((unsigned)E_ << 16);  // tensor_dim0[31:16]=0 | tensor_dim1[15:0]=E
  g1[3] = (int)(32u << 16);           // tensor_dim1[31:16]=0 | tile_dim0=32
  g1[4] = 64;                         // tile_dim1=64, tile_dim2=0
  g1[5] = E_;                         // tensor_dim0_stride[31:0]=E
  g1[6] = 0;                          // stride hi / tensor_dim1_stride lo
  g1[7] = 0;
  v4i z4 = {0, 0, 0, 0};
  v8i z8 = {0, 0, 0, 0, 0, 0, 0, 0};
  __builtin_amdgcn_tensor_load_to_lds(g0, g1, z4, z4, z8, 0);
}

// ---- f32 -> f16 conversion ----
__global__ void cvt_f32_f16_kernel(const float* __restrict__ in,
                                   half_t* __restrict__ out, int n) {
  int i = blockIdx.x * blockDim.x + threadIdx.x;
  if (i < n) out[i] = (half_t)in[i];
}

// ---- shared GEMM tile engine: C(128x64 per block) = A[M,K] * W[N,K]^T ----
// Block = 256 threads (8 waves). Wave w computes rows tileM+w*16, cols tileN..+63.
// W tile staged via TDM (tensor_load_to_lds), double-buffered: DMA of step i+1
// overlaps WMMA compute of step i; TENSORcnt is in-order so wait(1) => step i done.
static __device__ __forceinline__ void gemm_tile_accum(
    const half_t* __restrict__ A, const half_t* __restrict__ W,
    half_t* sW /* 2 buffers of 64*32 */, int tileM, int tileN, v8f acc[4]) {
  const int wave = threadIdx.x >> 5;
  const int rowBase = tileM + wave * 16;
  const int NSTEP = E_ / 32;
  if (wave == 0) tdm_issue_w_tile(W, tileN, 0, sW);
  for (int i = 0; i < NSTEP; ++i) {
    half_t* buf = sW + (i & 1) * (64 * 32);
    if (wave == 0) {
      if (i + 1 < NSTEP) {
        tdm_issue_w_tile(W, tileN, (i + 1) * 32, sW + ((i + 1) & 1) * (64 * 32));
        __builtin_amdgcn_s_wait_tensorcnt((short)1);  // step i complete
      } else {
        __builtin_amdgcn_s_wait_tensorcnt((short)0);  // last step complete
      }
    }
    __syncthreads();   // publish LDS tile i to all waves
    v16h a = load_frag_rowK(A + (size_t)rowBase * E_ + i * 32, E_);
#pragma unroll
    for (int t = 0; t < 4; ++t) {
      v16h b = load_fragB_colK(buf + t * 16 * 32, 32);
      acc[t] = wmma_f16(a, b, acc[t]);
    }
    __syncthreads();   // all reads of buf done before TDM overwrites it
  }
}

// QKV projection: output f16 in [B,H,S,D] layout (head-split).
__global__ __launch_bounds__(256) void proj_gemm_kernel(
    const half_t* __restrict__ A, const half_t* __restrict__ W,
    const float* __restrict__ bias, half_t* __restrict__ out) {
  __shared__ half_t sW[2 * 64 * 32];
  const int lane = threadIdx.x & 31;
  const int wave = threadIdx.x >> 5;
  const int tileM = blockIdx.x * 128;
  const int tileN = blockIdx.y * 64;
  v8f acc[4] = {{}, {}, {}, {}};
  gemm_tile_accum(A, W, sW, tileM, tileN, acc);
  const int hi = lane >> 4, lo = lane & 15;
  const int rowBase = tileM + wave * 16;
#pragma unroll
  for (int t = 0; t < 4; ++t) {
    const int col = tileN + t * 16 + lo;       // n = h*D + d
    const float bval = bias[col];
    const int h = col >> 6, d = col & (D_ - 1);
#pragma unroll
    for (int r = 0; r < 8; ++r) {
      const int row = rowBase + hi * 8 + r;    // m = b*S + s
      const int b = row >> 11, s = row & (S_ - 1);
      out[((size_t)(b * H_ + h) * S_ + s) * D_ + d] = (half_t)(acc[t][r] + bval);
    }
  }
}

// Output projection: attended[B*S,E] (f16) @ Wo^T + bo -> f32 d_out.
__global__ __launch_bounds__(256) void oproj_gemm_kernel(
    const half_t* __restrict__ A, const half_t* __restrict__ W,
    const float* __restrict__ bias, float* __restrict__ out) {
  __shared__ half_t sW[2 * 64 * 32];
  const int lane = threadIdx.x & 31;
  const int wave = threadIdx.x >> 5;
  const int tileM = blockIdx.x * 128;
  const int tileN = blockIdx.y * 64;
  v8f acc[4] = {{}, {}, {}, {}};
  gemm_tile_accum(A, W, sW, tileM, tileN, acc);
  const int hi = lane >> 4, lo = lane & 15;
  const int rowBase = tileM + wave * 16;
#pragma unroll
  for (int t = 0; t < 4; ++t) {
    const int col = tileN + t * 16 + lo;
    const float bval = bias[col];
#pragma unroll
    for (int r = 0; r < 8; ++r) {
      const int row = rowBase + hi * 8 + r;
      out[(size_t)row * E_ + col] = acc[t][r] + bval;
    }
  }
}

// ---- attention pass 1: per (b,h,q) row-max m and sum-exp l (online) ----
__global__ __launch_bounds__(32) void attn_stats_kernel(
    const half_t* __restrict__ Qh, const half_t* __restrict__ Kh,
    float* __restrict__ mBuf, float* __restrict__ lBuf) {
  const int bid = blockIdx.x;          // B*H*(S/16)
  const int qt = bid & 127;
  const int bh = bid >> 7;             // b*H + h
  const int lane = threadIdx.x & 31;
  const half_t* Qb = Qh + ((size_t)bh * S_ + qt * 16) * D_;
  const half_t* Kb = Kh + (size_t)bh * S_ * D_;
  const v16h a0 = load_frag_rowK(Qb, D_);        // d = 0..31
  const v16h a1 = load_frag_rowK(Qb + 32, D_);   // d = 32..63
  float m[8], l[8];
#pragma unroll
  for (int r = 0; r < 8; ++r) { m[r] = -1e30f; l[r] = 0.f; }
  for (int kt = 0; kt < S_ / 16; ++kt) {
    const half_t* kp = Kb + (size_t)kt * 16 * D_;
    if (kt + 2 < S_ / 16) __builtin_prefetch(Kb + (size_t)(kt + 2) * 16 * D_, 0, 0);
    v16h b0 = load_fragB_colK(kp, D_);
    v16h b1 = load_fragB_colK(kp + 32, D_);
    v8f c = {};
    c = wmma_f16(a0, b0, c);
    c = wmma_f16(a1, b1, c);
#pragma unroll
    for (int r = 0; r < 8; ++r) {
      float s = c[r] * SCALE_;
      float mn = fmaxf(m[r], s);
      l[r] = l[r] * __expf(m[r] - mn) + __expf(s - mn);
      m[r] = mn;
    }
  }
  // merge (m,l) across the 16 lanes that share each row
#pragma unroll
  for (int off = 1; off < 16; off <<= 1) {
#pragma unroll
    for (int r = 0; r < 8; ++r) {
      float mo = __shfl_xor(m[r], off);
      float lo = __shfl_xor(l[r], off);
      float mn = fmaxf(m[r], mo);
      l[r] = l[r] * __expf(m[r] - mn) + lo * __expf(mo - mn);
      m[r] = mn;
    }
  }
  if ((lane & 15) == 0) {
    const int rbase = qt * 16 + ((lane >> 4) << 3);
#pragma unroll
    for (int r = 0; r < 8; ++r) {
      mBuf[(size_t)bh * S_ + rbase + r] = m[r];
      lBuf[(size_t)bh * S_ + rbase + r] = l[r];
    }
  }
}

// ---- attention pass 2: O = (exp(S-m)/l) @ V, attended stored f16 [B,S,E] ----
__global__ __launch_bounds__(32) void attn_out_kernel(
    const half_t* __restrict__ Qh, const half_t* __restrict__ Kh,
    const half_t* __restrict__ Vh, const float* __restrict__ mBuf,
    const float* __restrict__ lBuf, half_t* __restrict__ attnOut) {
  __shared__ half_t sP[16 * 32];       // P tile, (q,k) row-major, k-pair wide
  const int bid = blockIdx.x;
  const int qt = bid & 127;
  const int bh = bid >> 7;
  const int lane = threadIdx.x & 31;
  const int hi = lane >> 4, lo = lane & 15;
  const half_t* Qb = Qh + ((size_t)bh * S_ + qt * 16) * D_;
  const half_t* Kb = Kh + (size_t)bh * S_ * D_;
  const half_t* Vb = Vh + (size_t)bh * S_ * D_;
  const v16h a0 = load_frag_rowK(Qb, D_);
  const v16h a1 = load_frag_rowK(Qb + 32, D_);
  float mrow[8], linv[8];
#pragma unroll
  for (int r = 0; r < 8; ++r) {
    const int row = qt * 16 + hi * 8 + r;
    mrow[r] = mBuf[(size_t)bh * S_ + row];
    linv[r] = 1.0f / lBuf[(size_t)bh * S_ + row];
  }
  v8f o[4] = {{}, {}, {}, {}};
  for (int kt2 = 0; kt2 < S_ / 32; ++kt2) {
    if (kt2 + 1 < S_ / 32) __builtin_prefetch(Vb + (size_t)(kt2 + 1) * 32 * D_, 0, 0);
    __syncthreads();
#pragma unroll
    for (int h2 = 0; h2 < 2; ++h2) {
      const int kt = kt2 * 2 + h2;
      const half_t* kp = Kb + (size_t)kt * 16 * D_;
      v16h b0 = load_fragB_colK(kp, D_);
      v16h b1 = load_fragB_colK(kp + 32, D_);
      v8f c = {};
      c = wmma_f16(a0, b0, c);
      c = wmma_f16(a1, b1, c);
#pragma unroll
      for (int r = 0; r < 8; ++r) {
        const int q = hi * 8 + r;
        float p = __expf(c[r] * SCALE_ - mrow[r]) * linv[r];
        sP[q * 32 + h2 * 16 + lo] = (half_t)p;
      }
    }
    __syncthreads();
    v16h pfrag = load_frag_rowK(&sP[0], 32);     // A-fragment of P (16x32)
#pragma unroll
    for (int dt = 0; dt < 4; ++dt) {
      v16h bv = load_fragB_rowK(Vb + (size_t)kt2 * 32 * D_ + dt * 16, D_);
      o[dt] = wmma_f16(pfrag, bv, o[dt]);
    }
  }
  const int b = bh >> 4, h = bh & (H_ - 1);
#pragma unroll
  for (int dt = 0; dt < 4; ++dt) {
#pragma unroll
    for (int r = 0; r < 8; ++r) {
      const int s = qt * 16 + hi * 8 + r;
      attnOut[((size_t)(b * S_ + s)) * E_ + h * D_ + dt * 16 + lo] =
          (half_t)o[dt][r];
    }
  }
}

// ---- attention pass 3: avg_attn[b,q,k] = mean_h P ----
__global__ __launch_bounds__(32) void attn_avg_kernel(
    const half_t* __restrict__ Qh, const half_t* __restrict__ Kh,
    const float* __restrict__ mBuf, const float* __restrict__ lBuf,
    float* __restrict__ avgOut) {
  int idx = blockIdx.x;                 // B * (S/16) * (S/16)
  const int kt = idx & 127; idx >>= 7;
  const int qt = idx & 127; idx >>= 7;
  const int b = idx;
  const int lane = threadIdx.x & 31;
  const int hi = lane >> 4, lo = lane & 15;
  float acc[8] = {0, 0, 0, 0, 0, 0, 0, 0};
  for (int h = 0; h < H_; ++h) {
    const int bh = b * H_ + h;
    const half_t* Qb = Qh + ((size_t)bh * S_ + qt * 16) * D_;
    const half_t* Kb = Kh + ((size_t)bh * S_ + kt * 16) * D_;
    v16h a0 = load_frag_rowK(Qb, D_);
    v16h a1 = load_frag_rowK(Qb + 32, D_);
    v16h b0 = load_fragB_colK(Kb, D_);
    v16h b1 = load_fragB_colK(Kb + 32, D_);
    v8f c = {};
    c = wmma_f16(a0, b0, c);
    c = wmma_f16(a1, b1, c);
#pragma unroll
    for (int r = 0; r < 8; ++r) {
      const int row = qt * 16 + hi * 8 + r;
      const float mm = mBuf[(size_t)bh * S_ + row];
      const float ll = lBuf[(size_t)bh * S_ + row];
      acc[r] += __expf(c[r] * SCALE_ - mm) / ll;
    }
  }
#pragma unroll
  for (int r = 0; r < 8; ++r) {
    const int q = qt * 16 + hi * 8 + r;
    avgOut[((size_t)b * S_ + q) * S_ + kt * 16 + lo] = acc[r] * (1.0f / H_);
  }
}

extern "C" void kernel_launch(void* const* d_in, const int* in_sizes, int n_in,
                              void* d_out, int out_size, void* d_ws,
                              size_t ws_size, hipStream_t stream) {
  (void)in_sizes; (void)n_in; (void)out_size; (void)ws_size;
  const float* x_q = (const float*)d_in[0];
  const float* x_k = (const float*)d_in[1];
  const float* x_v = (const float*)d_in[2];
  const float* Wq = (const float*)d_in[3];
  const float* bq = (const float*)d_in[4];
  const float* Wk = (const float*)d_in[5];
  const float* bk = (const float*)d_in[6];
  const float* Wv = (const float*)d_in[7];
  const float* bv = (const float*)d_in[8];
  const float* Wo = (const float*)d_in[9];
  const float* bo = (const float*)d_in[10];

  const int nX = B_ * S_ * E_;   // 4,194,304
  const int nW = E_ * E_;        // 1,048,576

  size_t off = 0;
  char* wsb = (char*)d_ws;
  auto take = [&](size_t bytes) -> void* {
    void* p = wsb + off;
    off += (bytes + 255) & ~(size_t)255;
    return p;
  };
  half_t* XqH = (half_t*)take((size_t)nX * 2);
  half_t* XkH = (half_t*)take((size_t)nX * 2);
  half_t* XvH = (half_t*)take((size_t)nX * 2);
  half_t* WqH = (half_t*)take((size_t)nW * 2);
  half_t* WkH = (half_t*)take((size_t)nW * 2);
  half_t* WvH = (half_t*)take((size_t)nW * 2);
  half_t* WoH = (half_t*)take((size_t)nW * 2);
  half_t* Qh = (half_t*)take((size_t)nX * 2);   // [B,H,S,D]
  half_t* Kh = (half_t*)take((size_t)nX * 2);
  half_t* Vh = (half_t*)take((size_t)nX * 2);
  float* mBuf = (float*)take((size_t)B_ * H_ * S_ * 4);
  float* lBuf = (float*)take((size_t)B_ * H_ * S_ * 4);
  half_t* attnH = (half_t*)take((size_t)nX * 2);  // attended, [B,S,E] f16

  float* out = (float*)d_out;
  float* avgOut = out + (size_t)B_ * S_ * E_;

  // 1) convert inputs & weights to f16
  const int CVT_B = 256;
  cvt_f32_f16_kernel<<<(nX + CVT_B - 1) / CVT_B, CVT_B, 0, stream>>>(x_q, XqH, nX);
  cvt_f32_f16_kernel<<<(nX + CVT_B - 1) / CVT_B, CVT_B, 0, stream>>>(x_k, XkH, nX);
  cvt_f32_f16_kernel<<<(nX + CVT_B - 1) / CVT_B, CVT_B, 0, stream>>>(x_v, XvH, nX);
  cvt_f32_f16_kernel<<<(nW + CVT_B - 1) / CVT_B, CVT_B, 0, stream>>>(Wq, WqH, nW);
  cvt_f32_f16_kernel<<<(nW + CVT_B - 1) / CVT_B, CVT_B, 0, stream>>>(Wk, WkH, nW);
  cvt_f32_f16_kernel<<<(nW + CVT_B - 1) / CVT_B, CVT_B, 0, stream>>>(Wv, WvH, nW);
  cvt_f32_f16_kernel<<<(nW + CVT_B - 1) / CVT_B, CVT_B, 0, stream>>>(Wo, WoH, nW);

  // 2) Q/K/V projections (WMMA GEMM + TDM double-buffered weight staging)
  dim3 gGemm(B_ * S_ / 128, E_ / 64);
  proj_gemm_kernel<<<gGemm, 256, 0, stream>>>(XqH, WqH, bq, Qh);
  proj_gemm_kernel<<<gGemm, 256, 0, stream>>>(XkH, WkH, bk, Kh);
  proj_gemm_kernel<<<gGemm, 256, 0, stream>>>(XvH, WvH, bv, Vh);

  // 3) softmax stats (flash pass 1)
  attn_stats_kernel<<<B_ * H_ * (S_ / 16), 32, 0, stream>>>(Qh, Kh, mBuf, lBuf);

  // 4) attended = P @ V (flash pass 2)
  attn_out_kernel<<<B_ * H_ * (S_ / 16), 32, 0, stream>>>(Qh, Kh, Vh, mBuf,
                                                          lBuf, attnH);

  // 5) avg_attn (head-mean of P), register accumulation, no atomics
  attn_avg_kernel<<<B_ * (S_ / 16) * (S_ / 16), 32, 0, stream>>>(Qh, Kh, mBuf,
                                                                 lBuf, avgOut);

  // 6) output projection -> f32 d_out
  oproj_gemm_kernel<<<gGemm, 256, 0, stream>>>(attnH, WoH, bo, out);
}
